// BasicAttention_86921548136884
// MI455X (gfx1250) — matrix operand order repaired and checked by reading, not compiled, not run on hardware
//
#include <hip/hip_runtime.h>

#define D 256
#define S 1024
#define NB 32
#define LDA 264              // padded LDS row stride (halves): 528B = 33*16B, keeps 16B alignment
#define kNEG (-1.0e30f)

typedef _Float16 half8 __attribute__((ext_vector_type(8)));
typedef _Float16 v16h  __attribute__((ext_vector_type(16)));
typedef float    v8f   __attribute__((ext_vector_type(8)));

#if defined(__has_builtin)
#if __has_builtin(__builtin_amdgcn_global_load_async_to_lds_b128)
#define HAVE_ASYNC 1
#endif
#endif

#ifdef HAVE_ASYNC
// Direct memory -> LDS async copy (16B per lane), tracked by ASYNCcnt; no VGPR round trip.
typedef int vi4 __attribute__((vector_size(16)));
typedef __attribute__((address_space(1))) vi4* as1_vi4_ptr;
typedef __attribute__((address_space(3))) vi4* as3_vi4_ptr;

__device__ __forceinline__ void async_cp16(void* lds, const void* g) {
  __builtin_amdgcn_global_load_async_to_lds_b128(
      (as1_vi4_ptr)(void*)g, (as3_vi4_ptr)lds, 0, 0);
}
__device__ __forceinline__ void wait_async0() {
#if __has_builtin(__builtin_amdgcn_s_wait_asynccnt)
  __builtin_amdgcn_s_wait_asynccnt(0);
#else
  asm volatile("s_wait_asynccnt 0" ::: "memory");
#endif
}
#endif

// Assemble a 16-half WMMA fragment from two 8-half LDS chunks (base + constant offsets).
__device__ __forceinline__ v16h mk_frag(const _Float16* p0, const _Float16* p1) {
  half8 lo = *(const half8*)p0;
  half8 hi = *(const half8*)p1;
  v16h out;
#pragma unroll
  for (int i = 0; i < 8; ++i) { out[i] = lo[i]; out[i + 8] = hi[i]; }
  return out;
}

// --- Kernel 0: W [d][n] f32 -> Wt [n][d] f16 -------------------------------
__global__ void k_wt(const float* __restrict__ W, _Float16* __restrict__ Wt) {
  int d = blockIdx.x;
  int n = threadIdx.x;
  Wt[n * D + d] = (_Float16)W[d * D + n];
}

// --- Kernel A: AW = f16(vecs) @ W -> f16, also emit f16 covecs -------------
__global__ void k_aw(const int* __restrict__ tokens,
                     const float* __restrict__ vec_table,
                     const float* __restrict__ covec_table,
                     const _Float16* __restrict__ Wt,
                     _Float16* __restrict__ AWh,
                     _Float16* __restrict__ Ch) {
  extern __shared__ char smem[];
  _Float16* Ws = (_Float16*)smem;        // 256 x LDA  (hot B-fragment buffer at LDS offset 0)
  _Float16* As = Ws + 256 * LDA;         // 128 x LDA
  __shared__ int toks[128];

  int qt = blockIdx.x, b = blockIdx.y;
  int tid = threadIdx.x;
  int q0 = qt * 128;

  if (tid < 128) toks[tid] = tokens[b * S + q0 + tid];

  // Wt (f16) -> LDS: 16B chunks, async path when available
#ifdef HAVE_ASYNC
  for (int e = tid; e < D * (D / 8); e += 256) {
    int row = e >> 5, cg = e & 31;
    async_cp16(Ws + row * LDA + cg * 8, Wt + row * D + cg * 8);
  }
#else
  for (int e = tid; e < D * (D / 8); e += 256) {
    int row = e >> 5, cg = e & 31;
    *(uint4*)(Ws + row * LDA + cg * 8) = *(const uint4*)(Wt + row * D + cg * 8);
  }
#endif
  __syncthreads();   // toks visible

  // gather vec rows -> LDS f16; covec rows -> global f16 (needs f32->f16 convert, stays VALU)
  for (int e = tid; e < 128 * D; e += 256) {
    int row = e >> 8, col = e & 255;
    long base = (long)toks[row] * D + col;
    As[row * LDA + col] = (_Float16)vec_table[base];
    Ch[(long)(b * S + q0 + row) * D + col] = (_Float16)covec_table[base];
  }
#ifdef HAVE_ASYNC
  wait_async0();
#endif
  __syncthreads();

  int wave = tid >> 5, lane = tid & 31;
  int l15 = lane & 15, h = lane >> 4;
  int qrow = wave * 16;

  // A-fragments: loaded once, outside the hot loop (large offsets harmless here)
  const _Float16* ap = As + (qrow + l15) * LDA + 8 * h;
  v16h a[8];
#pragma unroll
  for (int dk = 0; dk < 8; ++dk)
    a[dk] = mk_frag(ap + dk * 32, ap + dk * 32 + 16);

  // Output base: rows qb..qb+7 (stride D), columns ct*16+l15 (constant per ct)
  _Float16* awp = AWh + (long)(b * S + q0 + qrow + 8 * h) * D + l15;

  for (int ct = 0; ct < 16; ++ct) {
    // Ws at LDS offset 0: bp residual offsets dk*64 + {0,32} bytes fold into ds_load immediates
    const _Float16* bp = Ws + (ct * 16 + l15) * LDA + 8 * h;
    v8f acc = {};
#pragma unroll
    for (int dk = 0; dk < 8; ++dk) {
      v16h bfrag = mk_frag(bp + dk * 32, bp + dk * 32 + 16);
      acc = __builtin_amdgcn_wmma_f32_16x16x32_f16(false, a[dk], false, bfrag, (short)0, acc, false, false);
    }
#pragma unroll
    for (int r = 0; r < 8; ++r)
      awp[ct * 16 + r * D] = (_Float16)acc[r];
  }
}

// --- Kernel B: E tiles + fused masked row/col max --------------------------
__global__ void k_scores(const int* __restrict__ pads,
                         const _Float16* __restrict__ AWh,
                         const _Float16* __restrict__ Ch,
                         float* __restrict__ rowmax,
                         float* __restrict__ colpart) {
  extern __shared__ char smem[];
  _Float16* Cs = (_Float16*)smem;                // 128 x LDA (hot B-fragment buffer at LDS offset 0)
  _Float16* As = Cs + 128 * LDA;                 // 128 x LDA
  float* colbuf = (float*)(As + 128 * LDA);      // 8 waves x 128 cols

  int qt = blockIdx.x, b = blockIdx.y;
  int tid = threadIdx.x;
  int wave = tid >> 5, lane = tid & 31;
  int l15 = lane & 15, h = lane >> 4;
  int L = S - pads[b];
  int q0 = qt * 128;

  const _Float16* gA = AWh + (long)(b * S + q0) * D;
  for (int e = tid; e < 128 * (D / 8); e += 256) {
    int row = e >> 5, cg = e & 31;
#ifdef HAVE_ASYNC
    async_cp16(As + row * LDA + cg * 8, gA + row * D + cg * 8);
#else
    *(uint4*)(As + row * LDA + cg * 8) = *(const uint4*)(gA + row * D + cg * 8);
#endif
  }

  float rm[8];
#pragma unroll
  for (int r = 0; r < 8; ++r) rm[r] = kNEG;

  int qw = q0 + wave * 16 + 8 * h;
  const _Float16* ap = As + (wave * 16 + l15) * LDA + 8 * h;
  bool a_loaded = false;
  v16h a[8];

  for (int kblk = 0; kblk < 8; ++kblk) {
    __syncthreads();                             // Cs/colbuf reuse from previous iteration
    const _Float16* gC = Ch + (long)(b * S + kblk * 128) * D;
    for (int e = tid; e < 128 * (D / 8); e += 256) {
      int row = e >> 5, cg = e & 31;
#ifdef HAVE_ASYNC
      async_cp16(Cs + row * LDA + cg * 8, gC + row * D + cg * 8);
#else
      *(uint4*)(Cs + row * LDA + cg * 8) = *(const uint4*)(gC + row * D + cg * 8);
#endif
    }
#ifdef HAVE_ASYNC
    wait_async0();                               // also covers the A-tile copies on kblk==0
#endif
    __syncthreads();

    if (!a_loaded) {                             // A tile is stable across kblk iterations
      a_loaded = true;
#pragma unroll
      for (int dk = 0; dk < 8; ++dk) a[dk] = mk_frag(ap + dk * 32, ap + dk * 32 + 16);
    }

    for (int ct = 0; ct < 8; ++ct) {
      // Cs at LDS offset 0: fragment offsets fold into ds_load immediates
      const _Float16* bp = Cs + (ct * 16 + l15) * LDA + 8 * h;
      v8f acc = {};
#pragma unroll
      for (int dk = 0; dk < 8; ++dk) {
        v16h bfrag = mk_frag(bp + dk * 32, bp + dk * 32 + 16);
        acc = __builtin_amdgcn_wmma_f32_16x16x32_f16(false, a[dk], false, bfrag, (short)0, acc, false, false);
      }

      int kg = kblk * 128 + ct * 16 + l15;
      bool kvalid = kg < L;
      float cmax = kNEG;
#pragma unroll
      for (int r = 0; r < 8; ++r) {
        float v = (kvalid && (qw + r) < L) ? acc[r] : kNEG;
        rm[r] = fmaxf(rm[r], v);
        cmax = fmaxf(cmax, v);
      }
      cmax = fmaxf(cmax, __shfl_xor(cmax, 16));          // fold the two 8-row halves
      if ((lane & 16) == 0) colbuf[wave * 128 + ct * 16 + lane] = cmax;
    }
    __syncthreads();
    if (tid < 128) {
      float m = kNEG;
#pragma unroll
      for (int w = 0; w < 8; ++w) m = fmaxf(m, colbuf[w * 128 + tid]);
      colpart[((b * 8 + qt) << 10) + kblk * 128 + tid] = m;
    }
  }

#pragma unroll
  for (int r = 0; r < 8; ++r) {                          // reduce over 16 columns within half-wave
    float v = rm[r];
    v = fmaxf(v, __shfl_xor(v, 1));
    v = fmaxf(v, __shfl_xor(v, 2));
    v = fmaxf(v, __shfl_xor(v, 4));
    v = fmaxf(v, __shfl_xor(v, 8));
    if (l15 == 0) rowmax[(b << 10) + qw + r] = v;
  }
}

// --- Kernel C: softmaxes, weighted sums, linear head, log-softmax ----------
__global__ void k_final(const int* __restrict__ tokens,
                        const int* __restrict__ pads,
                        const float* __restrict__ vec_table,
                        const float* __restrict__ covec_table,
                        const float* __restrict__ lin_w,
                        const float* __restrict__ lin_b,
                        const float* __restrict__ rowmax,
                        const float* __restrict__ colpart,
                        float* __restrict__ out) {
  __shared__ float ak[S], aq[S];
  __shared__ float red[256];
  __shared__ int toks[S];
  __shared__ float X[2 * D];
  __shared__ float yv[8];

  int b = blockIdx.x, tid = threadIdx.x;
  int L = S - pads[b];

  for (int s = tid; s < S; s += 256) {
    float cm = kNEG;
#pragma unroll
    for (int qt = 0; qt < 8; ++qt) cm = fmaxf(cm, colpart[((b * 8 + qt) << 10) + s]);
    ak[s] = (s < L) ? cm : kNEG;
    aq[s] = (s < L) ? rowmax[(b << 10) + s] : kNEG;
    toks[s] = tokens[b * S + s];
  }
  __syncthreads();

  for (int which = 0; which < 2; ++which) {
    float* arr = which ? aq : ak;
    float lm = kNEG;
    for (int s = tid; s < S; s += 256) lm = fmaxf(lm, arr[s]);
    red[tid] = lm; __syncthreads();
    for (int o = 128; o > 0; o >>= 1) { if (tid < o) red[tid] = fmaxf(red[tid], red[tid + o]); __syncthreads(); }
    float mx = red[0]; __syncthreads();
    float ls = 0.f;
    for (int s = tid; s < S; s += 256) {
      float e = (s < L) ? __expf(arr[s] - mx) : 0.f;
      arr[s] = e; ls += e;
    }
    red[tid] = ls; __syncthreads();
    for (int o = 128; o > 0; o >>= 1) { if (tid < o) red[tid] += red[tid + o]; __syncthreads(); }
    float inv = 1.f / red[0]; __syncthreads();
    for (int s = tid; s < S; s += 256) arr[s] *= inv;
    __syncthreads();
  }

  {
    int d = tid;                          // 256 threads == D
    float av = 0.f, ac = 0.f;
    for (int s = 0; s < L; ++s) {
      long base = (long)toks[s] * D + d;
      av += ak[s] * vec_table[base];
      ac += aq[s] * covec_table[base];
    }
    X[d] = av; X[D + d] = ac;
  }
  __syncthreads();

  if (tid < 5) {
    float y = lin_b[tid];
    for (int j = 0; j < 2 * D; ++j) y += lin_w[tid * 2 * D + j] * X[j];
    yv[tid] = y;
  }
  __syncthreads();
  if (tid == 0) {
    float mx = yv[0];
    for (int c = 1; c < 5; ++c) mx = fmaxf(mx, yv[c]);
    float se = 0.f;
    for (int c = 0; c < 5; ++c) se += __expf(yv[c] - mx);
    float lse = mx + __logf(se);
    for (int c = 0; c < 5; ++c) out[b * 5 + c] = yv[c] - lse;
  }
}

extern "C" void kernel_launch(void* const* d_in, const int* in_sizes, int n_in,
                              void* d_out, int out_size, void* d_ws, size_t ws_size,
                              hipStream_t stream) {
  const int*   tokens      = (const int*)d_in[0];
  const int*   pads        = (const int*)d_in[1];
  const float* vec_table   = (const float*)d_in[2];
  const float* covec_table = (const float*)d_in[3];
  const float* W           = (const float*)d_in[4];
  const float* lin_w       = (const float*)d_in[5];
  const float* lin_b       = (const float*)d_in[6];
  float* out = (float*)d_out;

  char* ws = (char*)d_ws;
  _Float16* Wt      = (_Float16*)ws;                                   // 128 KB
  _Float16* AWh     = (_Float16*)(ws + (131072));                      // 16 MB
  _Float16* Ch      = (_Float16*)(ws + (131072 + 16777216));           // 16 MB
  float*    rowmax  = (float*)  (ws + (131072 + 2 * 16777216));        // 128 KB
  float*    colpart = (float*)  (ws + (131072 + 2 * 16777216 + 131072)); // 1 MB

  k_wt<<<dim3(256), dim3(256), 0, stream>>>(W, Wt);

  size_t smemA = (size_t)(128 * LDA + 256 * LDA) * sizeof(_Float16);
  k_aw<<<dim3(8, NB), dim3(256), smemA, stream>>>(tokens, vec_table, covec_table, Wt, AWh, Ch);

  size_t smemB = (size_t)(2 * 128 * LDA) * sizeof(_Float16) + 8 * 128 * sizeof(float);
  k_scores<<<dim3(8, NB), dim3(256), smemB, stream>>>(pads, AWh, Ch, rowmax, colpart);

  k_final<<<dim3(NB), dim3(256), 0, stream>>>(tokens, pads, vec_table, covec_table,
                                              lin_w, lin_b, rowmax, colpart, out);
}
